// IncoherentSdpaAttention_6030134083826
// MI455X (gfx1250) — compile-verified, matrix operand order checked by reading, not run on hardware
//
#include <hip/hip_runtime.h>
#include <math.h>

// ---------------------------------------------------------------------------
// IncoherentSdpaAttention for MI455X (gfx1250), wave32 + WMMA bf16
// ---------------------------------------------------------------------------

typedef __bf16 bf16_t;
typedef __attribute__((ext_vector_type(16))) __bf16 v16bf;
typedef __attribute__((ext_vector_type(8)))  float  v8f;

#define HID     4096
#define NH      32
#define NKV     8
#define HD      128
#define KV_OUT  1024
#define SEQ     2048
#define SCALE_F 64.0f
#define THETA_F 500000.0f

// float -> bf16 round-to-nearest-even (bit-level)
__device__ __forceinline__ bf16_t f2bf(float f) {
  union { float f; unsigned u; } a; a.f = f;
  unsigned r = a.u + 0x7FFFu + ((a.u >> 16) & 1u);
  union { unsigned short s; bf16_t b; } o; o.s = (unsigned short)(r >> 16);
  return o.b;
}

union ABFrag { v16bf v; float4 f4[2]; bf16_t e[16]; };

// 16x32 bf16 A/B fragment from a row-major MxK / NxK tile (ISA 7.12.2 layout):
// lanes 0-15 row=lane, K={k0..k0+7, k0+16..k0+23}; lanes 16-31 K shifted by 8.
__device__ __forceinline__ v16bf load_ab_frag(const bf16_t* __restrict__ rowptr,
                                              int k0, int half) {
  ABFrag f;
  f.f4[0] = *reinterpret_cast<const float4*>(rowptr + k0 + half * 8);
  f.f4[1] = *reinterpret_cast<const float4*>(rowptr + k0 + 16 + half * 8);
  return f.v;
}

// ---------------------------------------------------------------------------
// Kernel 1: sign-flip + normalized FWHT(4096) + 1/SCALE, emit bf16
// ---------------------------------------------------------------------------
__global__ __launch_bounds__(1024) void fwht_kernel(const float* __restrict__ in,
                                                    const float* __restrict__ su,
                                                    bf16_t* __restrict__ out) {
  __shared__ float buf[HID];
  const int row = blockIdx.x;
  const float* x = in + (size_t)row * HID;
  for (int i = threadIdx.x; i < HID; i += blockDim.x) buf[i] = x[i] * su[i];
  __syncthreads();
  for (int h = 1; h < HID; h <<= 1) {
    for (int p = threadIdx.x; p < HID / 2; p += blockDim.x) {
      int lo  = p & (h - 1);
      int idx = ((p ^ lo) << 1) | lo;           // (p/h)*2h + (p%h)
      float a = buf[idx], b = buf[idx + h];
      buf[idx]     = a + b;
      buf[idx + h] = a - b;
    }
    __syncthreads();
  }
  const float norm = 1.0f / (64.0f * SCALE_F);  // 1/sqrt(4096) / SCALE
  bf16_t* o = out + (size_t)row * HID;
  for (int i = threadIdx.x; i < HID; i += blockDim.x) o[i] = f2bf(buf[i] * norm);
}

// ---------------------------------------------------------------------------
// Kernel 2: fp32 -> bf16 weight conversion
// ---------------------------------------------------------------------------
__global__ void f2bf_kernel(const float* __restrict__ in, bf16_t* __restrict__ out, int n) {
  int i = blockIdx.x * blockDim.x + threadIdx.x;
  if (i < n) out[i] = f2bf(in[i]);
}

// ---------------------------------------------------------------------------
// Kernel 3: C[M][N] = (A[MxK] @ Bt[NxK]^T) * scale[n] * smul
// Block tile 64(M) x 128(N), 8 waves in a 2x4 grid, 32x32 (2x2 WMMA) per wave.
// A tiles (64x32 / k-step) staged cooperatively into LDS with
// global_load_async_to_lds_b128, double-buffered (ASYNCcnt path).
// B streams directly from global (weights are L2-resident) with prefetch.
// grid = (N/128, M/64), block = 256.  K multiple of 32, preload guarded.
// ---------------------------------------------------------------------------
__global__ __launch_bounds__(256) void gemm_bf16_kernel(
    const bf16_t* __restrict__ A, const bf16_t* __restrict__ Bt,
    const float* __restrict__ scale, float* __restrict__ C,
    int M, int N, int K, float smul) {
  __shared__ bf16_t abuf[2][64][32];   // 2 x 4KB double buffer
  const int tid    = threadIdx.x;
  const int lane   = tid & 31;
  const int wave   = tid >> 5;
  const int wave_m = wave >> 2;        // 0..1  (32-row half of 64)
  const int wave_n = wave & 3;         // 0..3  (32-col quarter of 128)
  const int mb = blockIdx.y * 64;
  const int nb = blockIdx.x * 128;
  const int mr   = lane & 15;
  const int half = lane >> 4;

  // async staging assignment: thread -> (row 0..63, 16B segment 0..3)
  const int srow = tid >> 2;
  const int sseg = tid & 3;
  const bf16_t* gsrc = A + (size_t)(mb + srow) * K + sseg * 8;
  const unsigned lds0 = (unsigned)(uintptr_t)&abuf[0][srow][sseg * 8];
  const unsigned lds1 = (unsigned)(uintptr_t)&abuf[1][srow][sseg * 8];

  const bf16_t* brow0 = Bt + (size_t)(nb + wave_n * 32 + mr) * K;
  const bf16_t* brow1 = Bt + (size_t)(nb + wave_n * 32 + 16 + mr) * K;

  v8f acc00 = {}, acc01 = {}, acc10 = {}, acc11 = {};

  // preload k=0 tile into buffer 0
  asm volatile("global_load_async_to_lds_b128 %0, %1, off"
               :: "v"(lds0), "v"((unsigned long long)(uintptr_t)gsrc)
               : "memory");
  asm volatile("s_wait_asynccnt 0x0" ::: "memory");
  __syncthreads();

  int cur = 0;
  for (int k0 = 0; k0 < K; k0 += 32) {
    if (k0 + 32 < K) {                 // issue next A tile (other buffer)
      unsigned ldsn = cur ? lds0 : lds1;
      asm volatile("global_load_async_to_lds_b128 %0, %1, off"
                   :: "v"(ldsn),
                      "v"((unsigned long long)(uintptr_t)(gsrc + k0 + 32))
                   : "memory");
    }
    __builtin_prefetch(brow0 + k0 + 256, 0, 1);
    __builtin_prefetch(brow1 + k0 + 256, 0, 1);

    v16bf a0 = load_ab_frag(&abuf[cur][wave_m * 32 + mr][0], 0, half);
    v16bf a1 = load_ab_frag(&abuf[cur][wave_m * 32 + 16 + mr][0], 0, half);
    v16bf b0 = load_ab_frag(brow0, k0, half);
    v16bf b1 = load_ab_frag(brow1, k0, half);
    acc00 = __builtin_amdgcn_wmma_f32_16x16x32_bf16(false, a0, false, b0, (short)0, acc00, false, false);
    acc01 = __builtin_amdgcn_wmma_f32_16x16x32_bf16(false, a0, false, b1, (short)0, acc01, false, false);
    acc10 = __builtin_amdgcn_wmma_f32_16x16x32_bf16(false, a1, false, b0, (short)0, acc10, false, false);
    acc11 = __builtin_amdgcn_wmma_f32_16x16x32_bf16(false, a1, false, b1, (short)0, acc11, false, false);

    asm volatile("s_wait_asynccnt 0x0" ::: "memory");
    __syncthreads();
    cur ^= 1;
  }

  // epilogue: D layout lane -> (m = v + 8*half, n = mr) per 16x16 sub-tile
  const int ncol = nb + wave_n * 32 + mr;
  const float sc0 = scale[ncol] * smul;
  const float sc1 = scale[ncol + 16] * smul;
#pragma unroll
  for (int mi = 0; mi < 2; ++mi) {
    float* crow = C + (size_t)(mb + wave_m * 32 + mi * 16 + 8 * half) * N + ncol;
    const v8f ac0 = mi ? acc10 : acc00;
    const v8f ac1 = mi ? acc11 : acc01;
#pragma unroll
    for (int v = 0; v < 8; ++v) {
      crow[(size_t)v * N]      = ac0[v] * sc0;
      crow[(size_t)v * N + 16] = ac1[v] * sc1;
    }
  }
}

// ---------------------------------------------------------------------------
// Kernel 4: RoPE + repack [s][nh*HD] fp32 -> [h][s][HD] bf16
// ---------------------------------------------------------------------------
__global__ void rope_pack_kernel(const float* __restrict__ xf, bf16_t* __restrict__ out,
                                 int nheads) {
  int idx = blockIdx.x * blockDim.x + threadIdx.x;
  int total = SEQ * nheads * HD;
  if (idx >= total) return;
  int d = idx % HD;
  int h = (idx / HD) % nheads;
  int s = idx / (HD * nheads);
  const float* row = xf + (size_t)s * nheads * HD + h * HD;
  float v = row[d];
  int j = d & 63;
  float inv = expf(-(float)j * (logf(THETA_F) / 64.0f));
  float ang = (float)s * inv;
  float c = cosf(ang), sn = sinf(ang);
  int   dp = (d < 64) ? d + 64 : d - 64;
  float vp = row[dp];
  float r  = (d < 64) ? (v * c - vp * sn) : (v * c + vp * sn);
  out[((size_t)h * SEQ + s) * HD + d] = f2bf(r);
}

// ---------------------------------------------------------------------------
// Kernel 5: V repack [s][NKV*HD] fp32 -> transposed [hkv][HD][SEQ] bf16
// ---------------------------------------------------------------------------
__global__ void vtrans_kernel(const float* __restrict__ vf, bf16_t* __restrict__ vT) {
  int idx = blockIdx.x * blockDim.x + threadIdx.x;
  if (idx >= SEQ * KV_OUT) return;
  int n = idx % KV_OUT;          // h*HD + d
  int s = idx / KV_OUT;
  vT[(size_t)n * SEQ + s] = f2bf(vf[idx]);
}

// ---------------------------------------------------------------------------
// Kernel 6: causal flash attention (GQA 32/8, HD=128)
// grid = (SEQ/128, NH), block = 256; each wave owns 16 query rows.
// Per 32-key iter: 8 WMMA (S) -> online softmax (fp32, shfl_xor row-reduce)
//   -> P via per-wave LDS re-fragmentation -> 8 WMMA (O accum).
// No __syncthreads: waves have different causal trip counts.
// ---------------------------------------------------------------------------
__global__ __launch_bounds__(256) void flash_attn_kernel(
    const bf16_t* __restrict__ Q,    // [NH][SEQ][HD]
    const bf16_t* __restrict__ Kc,   // [NKV][SEQ][HD]
    const bf16_t* __restrict__ Vt,   // [NKV][HD][SEQ]
    float* __restrict__ O) {         // [SEQ][NH*HD]
  __shared__ float sbuf[8][16][32];
  const int lane = threadIdx.x & 31;
  const int wave = threadIdx.x >> 5;
  const int h    = blockIdx.y;
  const int hkv  = h / (NH / NKV);
  const int q0   = (blockIdx.x * 8 + wave) * 16;
  const int mr   = lane & 15;
  const int half = lane >> 4;

  const bf16_t* qrow = Q + ((size_t)h * SEQ + q0 + mr) * HD;
  v16bf qf[4];
#pragma unroll
  for (int c = 0; c < 4; ++c) qf[c] = load_ab_frag(qrow, c * 32, half);

  v8f o_acc[8];
#pragma unroll
  for (int t = 0; t < 8; ++t) o_acc[t] = {};
  float m_i[8], l_i[8];
#pragma unroll
  for (int v = 0; v < 8; ++v) { m_i[v] = -1e30f; l_i[v] = 0.0f; }

  const float sm_scale = 0.088388347648318447f;  // 1/sqrt(128)
  const int kv_end = q0 + 16;                    // causal bound (exclusive)

  for (int kt = 0; kt < kv_end; kt += 32) {
    // ---- S = Q @ K^T : two 16-key N-tiles, K-dim = HD in 4 chunks
    v8f s_acc[2];
#pragma unroll
    for (int nt = 0; nt < 2; ++nt) {
      const bf16_t* krow = Kc + ((size_t)hkv * SEQ + kt + nt * 16 + mr) * HD;
      v8f a = {};
#pragma unroll
      for (int c = 0; c < 4; ++c) {
        v16bf kfr = load_ab_frag(krow, c * 32, half);
        a = __builtin_amdgcn_wmma_f32_16x16x32_bf16(false, qf[c], false, kfr,
                                                    (short)0, a, false, false);
      }
      s_acc[nt] = a;
    }
    // ---- scale + causal mask (D layout: row m=v+8*half, key n=kt+nt*16+mr)
#pragma unroll
    for (int nt = 0; nt < 2; ++nt) {
      int kg = kt + nt * 16 + mr;
#pragma unroll
      for (int v = 0; v < 8; ++v) {
        int qg = q0 + v + 8 * half;
        float sv = s_acc[nt][v] * sm_scale;
        s_acc[nt][v] = (kg <= qg) ? sv : -1e30f;
      }
    }
    // ---- online softmax
#pragma unroll
    for (int v = 0; v < 8; ++v) {
      float mx = fmaxf(s_acc[0][v], s_acc[1][v]);
#pragma unroll
      for (int off = 1; off < 16; off <<= 1) mx = fmaxf(mx, __shfl_xor(mx, off, 32));
      float mnew  = fmaxf(m_i[v], mx);
      float alpha = expf(m_i[v] - mnew);
      m_i[v] = mnew;
      float p0 = expf(s_acc[0][v] - mnew);
      float p1 = expf(s_acc[1][v] - mnew);
      s_acc[0][v] = p0; s_acc[1][v] = p1;
      float rs = p0 + p1;
#pragma unroll
      for (int off = 1; off < 16; off <<= 1) rs += __shfl_xor(rs, off, 32);
      l_i[v] = l_i[v] * alpha + rs;
#pragma unroll
      for (int t = 0; t < 8; ++t) o_acc[t][v] *= alpha;
    }
    // ---- D-layout -> A-layout re-fragmentation of P via per-wave LDS
#pragma unroll
    for (int nt = 0; nt < 2; ++nt)
#pragma unroll
      for (int v = 0; v < 8; ++v)
        sbuf[wave][v + 8 * half][nt * 16 + mr] = s_acc[nt][v];
    ABFrag pf;
#pragma unroll
    for (int j = 0; j < 8; ++j) {
      pf.e[j]     = f2bf(sbuf[wave][mr][half * 8 + j]);
      pf.e[8 + j] = f2bf(sbuf[wave][mr][16 + half * 8 + j]);
    }
    // ---- O += P @ V : 8 d-tiles, B rows from transposed V cache
#pragma unroll
    for (int t = 0; t < 8; ++t) {
      const bf16_t* vrow = Vt + ((size_t)hkv * HD + t * 16 + mr) * SEQ;
      v16bf vfr = load_ab_frag(vrow, kt, half);
      o_acc[t] = __builtin_amdgcn_wmma_f32_16x16x32_bf16(false, pf.v, false, vfr,
                                                         (short)0, o_acc[t], false, false);
    }
  }
  // ---- finalize: O /= l, store [s][NH*HD]
#pragma unroll
  for (int t = 0; t < 8; ++t)
#pragma unroll
    for (int v = 0; v < 8; ++v) {
      int m = v + 8 * half;
      O[(size_t)(q0 + m) * (NH * HD) + h * HD + t * 16 + mr] = o_acc[t][v] / l_i[v];
    }
}

// ---------------------------------------------------------------------------
// Host-side orchestration
// ---------------------------------------------------------------------------
extern "C" void kernel_launch(void* const* d_in, const int* in_sizes, int n_in,
                              void* d_out, int out_size, void* d_ws, size_t ws_size,
                              hipStream_t stream) {
  (void)in_sizes; (void)n_in; (void)out_size; (void)ws_size;
  const float* hidden     = (const float*)d_in[0];
  /* position_ids (d_in[1]) is arange(S); computed inline */
  const float* SU_qkv     = (const float*)d_in[2];
  const float* SU_o       = (const float*)d_in[3];
  const float* Wscale_qkv = (const float*)d_in[4];
  const float* Wscale_o   = (const float*)d_in[5];
  const float* Wq         = (const float*)d_in[6];
  const float* Wk         = (const float*)d_in[7];
  const float* Wv         = (const float*)d_in[8];
  const float* Wo         = (const float*)d_in[9];
  float* out = (float*)d_out;

  // workspace carve-out (256B aligned)
  char* ws = (char*)d_ws;
  size_t off = 0;
  auto carve = [&](size_t bytes) {
    void* p = ws + off;
    off = (off + bytes + 255) & ~(size_t)255;
    return p;
  };
  bf16_t* Wq_bf = (bf16_t*)carve((size_t)HID * HID * 2);
  bf16_t* Wk_bf = (bf16_t*)carve((size_t)KV_OUT * HID * 2);
  bf16_t* Wv_bf = (bf16_t*)carve((size_t)KV_OUT * HID * 2);
  bf16_t* Wo_bf = (bf16_t*)carve((size_t)HID * HID * 2);
  bf16_t* x_bf  = (bf16_t*)carve((size_t)SEQ * HID * 2);   // reused as y_bf
  float*  tmpf  = (float*) carve((size_t)SEQ * HID * 4);   // qf32, later attn-out
  float*  kf32  = (float*) carve((size_t)SEQ * KV_OUT * 4);
  float*  vf32  = (float*) carve((size_t)SEQ * KV_OUT * 4);
  bf16_t* q_bf  = (bf16_t*)carve((size_t)SEQ * HID * 2);
  bf16_t* k_bf  = (bf16_t*)carve((size_t)SEQ * KV_OUT * 2);
  bf16_t* vT_bf = (bf16_t*)carve((size_t)SEQ * KV_OUT * 2);

  // 1) weights -> bf16
  {
    int n1 = HID * HID, n2 = KV_OUT * HID;
    f2bf_kernel<<<(n1 + 255) / 256, 256, 0, stream>>>(Wq, Wq_bf, n1);
    f2bf_kernel<<<(n2 + 255) / 256, 256, 0, stream>>>(Wk, Wk_bf, n2);
    f2bf_kernel<<<(n2 + 255) / 256, 256, 0, stream>>>(Wv, Wv_bf, n2);
    f2bf_kernel<<<(n1 + 255) / 256, 256, 0, stream>>>(Wo, Wo_bf, n1);
  }
  // 2) x = fwht(hidden * SU_qkv) / SCALE  (bf16)
  fwht_kernel<<<SEQ, 1024, 0, stream>>>(hidden, SU_qkv, x_bf);

  // 3) QKV projections (epilogue: * Wscale * SCALE); grid = (N/128, M/64)
  gemm_bf16_kernel<<<dim3(HID / 128, SEQ / 64), 256, 0, stream>>>(
      x_bf, Wq_bf, Wscale_qkv, tmpf, SEQ, HID, HID, SCALE_F);
  gemm_bf16_kernel<<<dim3(KV_OUT / 128, SEQ / 64), 256, 0, stream>>>(
      x_bf, Wk_bf, Wscale_qkv + HID, kf32, SEQ, KV_OUT, HID, SCALE_F);
  gemm_bf16_kernel<<<dim3(KV_OUT / 128, SEQ / 64), 256, 0, stream>>>(
      x_bf, Wv_bf, Wscale_qkv + HID + KV_OUT, vf32, SEQ, KV_OUT, HID, SCALE_F);

  // 4) RoPE + head-major repack; V transpose
  {
    int nq = SEQ * NH * HD, nk = SEQ * NKV * HD, nv = SEQ * KV_OUT;
    rope_pack_kernel<<<(nq + 255) / 256, 256, 0, stream>>>(tmpf, q_bf, NH);
    rope_pack_kernel<<<(nk + 255) / 256, 256, 0, stream>>>(kf32, k_bf, NKV);
    vtrans_kernel<<<(nv + 255) / 256, 256, 0, stream>>>(vf32, vT_bf);
  }
  // 5) causal flash attention -> tmpf as [s][NH*HD]
  flash_attn_kernel<<<dim3(SEQ / 128, NH), 256, 0, stream>>>(q_bf, k_bf, vT_bf, tmpf);

  // 6) y2 = fwht(attn * SU_o) / SCALE (bf16, reuse x_bf)
  fwht_kernel<<<SEQ, 1024, 0, stream>>>(tmpf, SU_o, x_bf);

  // 7) out = y2 @ Wo^T * Wscale_o * SCALE
  gemm_bf16_kernel<<<dim3(HID / 128, SEQ / 64), 256, 0, stream>>>(
      x_bf, Wo_bf, Wscale_o, out, SEQ, HID, HID, SCALE_F);
}
